// WKS_2319282340282
// MI455X (gfx1250) — matrix operand order repaired and checked by reading, not compiled, not run on hardware
//
#include <hip/hip_runtime.h>
#include <hip/hip_bf16.h>
#include <math.h>

// WKS descriptors on MI455X (gfx1250).
// Core op: batched GEMM (M=128, K=200, N=100000, B=4) via v_wmma_f32_16x16x32_bf16.
// Memory-bound target ~40us: evecs read once (320MB), out written once + rescaled.

typedef __attribute__((ext_vector_type(16))) __bf16 v16bf;
typedef __attribute__((ext_vector_type(8)))  float  v8f;

#define ND    128          // n_descr
#define BATCH 4
#define KPAD  224          // K padded to 7*32 for bf16 WMMA K-blocks
#define NKB   7
#define EPSF  1e-12f

__device__ __forceinline__ unsigned short f32_to_bf16(float f) {
    unsigned int u = __float_as_uint(f);
    u += 0x7FFFu + ((u >> 16) & 1u);        // round to nearest even
    return (unsigned short)(u >> 16);
}

// ---------------------------------------------------------------------------
// Kernel 1: per-batch prep (sort evals, build bf16 coefs, inv scaling, zero norms)
// ---------------------------------------------------------------------------
__global__ void wks_prep(const float* __restrict__ evals,
                         unsigned short* __restrict__ coefsb,
                         float* __restrict__ sscale,
                         float* __restrict__ normacc,
                         int K) {
    __shared__ float s_abs[KPAD];
    __shared__ float s_srt[KPAD];
    const int t = threadIdx.x;

    for (int b = 0; b < BATCH; ++b) {
        if (t < K) s_abs[t] = fabsf(evals[b * K + t]);
        __syncthreads();
        // rank sort (K=200, trivial)
        if (t < K) {
            float x = s_abs[t];
            int r = 0;
            for (int j = 0; j < K; ++j) {
                float y = s_abs[j];
                r += (y < x) || (y == x && j < t);
            }
            s_srt[r] = x;
        }
        __syncthreads();
        float emin = logf(fmaxf(s_srt[1], EPSF));
        float emax = logf(fmaxf(s_srt[K - 1], EPSF));
        float sigma = 7.0f * (emax - emin) / (float)ND;
        emin += 2.0f * sigma;
        emax -= 2.0f * sigma;
        const float inv2s2 = 1.0f / (2.0f * sigma * sigma);
        __syncthreads();
        if (t < K) s_srt[t] = logf(fmaxf(s_srt[t], EPSF));   // log_ev in place
        __syncthreads();
        if (t < ND) {
            const float step   = (float)t / (float)(ND - 1);
            const float energy = emin * (1.0f - step) + emax * step;
            float sum = 0.0f;
            unsigned short* row = coefsb + (size_t)(b * ND + t) * KPAD;
            for (int k = 0; k < KPAD; ++k) {
                float cc = 0.0f;
                if (k < K) {
                    float d = energy - s_srt[k];
                    cc = expf(-d * d * inv2s2);
                    sum += cc;
                }
                row[k] = f32_to_bf16(cc);
            }
            sscale[b * ND + t]  = 1.0f / (sum + EPSF);
            normacc[b * ND + t] = 0.0f;     // re-zero every call (graph replay safe)
        }
        __syncthreads();
    }
}

// ---------------------------------------------------------------------------
// Kernel 2: GEMM wks = coefs @ evecs^2^T, scale by 1/inv_scaling, store (b,v,n),
//           accumulate feat_norm partials.
// 8 waves/WG; each wave: 16 columns (v), full M=128 -> 8 16x16 f32 accumulators.
// ---------------------------------------------------------------------------
__global__ void __launch_bounds__(256)
wks_gemm(const float* __restrict__ evecs,
         const float* __restrict__ mass,
         const unsigned short* __restrict__ coefsb,
         const float* __restrict__ sscale,
         float* __restrict__ normacc,
         float* __restrict__ out,
         int V, int K) {
    __shared__ __align__(16) unsigned short sA[ND * KPAD];   // 57344 B bf16 coefs
    __shared__ float snorm[8][ND];                           // 4096 B partials

    const int b    = blockIdx.y;
    const int tid  = threadIdx.x;
    const int wave = tid >> 5;
    const int lane = tid & 31;
    const int c    = lane & 15;   // column-in-tile / M-row-in-tile index
    const int hi   = lane >> 4;   // half-wave select

    // stage this batch's coefs into LDS (uint4 = 8 bf16 per load)
    {
        const uint4* src = reinterpret_cast<const uint4*>(coefsb + (size_t)b * ND * KPAD);
        uint4* dst = reinterpret_cast<uint4*>(sA);
        const int nq = ND * KPAD / 8;   // 3584
        for (int i = tid; i < nq; i += 256) dst[i] = src[i];
    }
    __syncthreads();

    const int v      = blockIdx.x * 128 + wave * 16 + c;
    const bool vok   = (v < V);
    const float mv   = vok ? mass[(size_t)b * V + v] : 0.0f;
    const float* brow = evecs + ((size_t)b * V + (size_t)(vok ? v : 0)) * (size_t)K;

    v8f acc[8] = {};

    for (int kb = 0; kb < NKB; ++kb) {
        const int k0 = kb * 32 + hi * 16;   // this lane's 16 contiguous K values
        float f[16];
        if (vok && (k0 + 16 <= K)) {
            const float4* p = reinterpret_cast<const float4*>(brow + k0);
            float4 q0 = p[0], q1 = p[1], q2 = p[2], q3 = p[3];
            f[0]=q0.x; f[1]=q0.y; f[2]=q0.z; f[3]=q0.w;
            f[4]=q1.x; f[5]=q1.y; f[6]=q1.z; f[7]=q1.w;
            f[8]=q2.x; f[9]=q2.y; f[10]=q2.z; f[11]=q2.w;
            f[12]=q3.x; f[13]=q3.y; f[14]=q3.z; f[15]=q3.w;
        } else {
#pragma unroll
            for (int i = 0; i < 16; ++i) {
                int k = k0 + i;
                f[i] = (vok && k < K) ? brow[k] : 0.0f;
            }
        }
        if (vok && (kb + 1 < NKB))
            __builtin_prefetch(brow + (kb + 1) * 32, 0, 3);   // global_prefetch_b8

        // B fragment (32x16 K x N bf16): lane holds col n=c, K = kb*32 + hi*16 + [0..15]
        union { v16bf vec; unsigned short u[16]; } bf;
#pragma unroll
        for (int i = 0; i < 16; ++i) bf.u[i] = f32_to_bf16(f[i] * f[i]);

#pragma unroll
        for (int t = 0; t < 8; ++t) {
            // A fragment (16x32 M x K bf16): lane row m = 16t + c, kbase = hi*16,
            // elements 0..15 -> k = kb*32 + kbase + [0..15] (contiguous)
            const int base = (t * 16 + c) * KPAD + kb * 32 + hi * 16;
            union { v16bf vec; uint4 q[2]; } af;
            af.q[0] = *reinterpret_cast<const uint4*>(&sA[base]);
            af.q[1] = *reinterpret_cast<const uint4*>(&sA[base + 8]);
            acc[t] = __builtin_amdgcn_wmma_f32_16x16x32_bf16(
                false, af.vec, false, bf.vec, (short)0, acc[t], false, false);
        }
    }

    // epilogue: D layout — lane (c,hi), reg r of tile t -> n_d = 16t + 8*hi + r, col v
    const float* ss = sscale + b * ND;
#pragma unroll
    for (int t = 0; t < 8; ++t) {
        float vals[8];
#pragma unroll
        for (int r = 0; r < 8; ++r) {
            const int nd = t * 16 + hi * 8 + r;
            float fe = acc[t][r] * ss[nd];
            vals[r] = fe;
            float p = fe * fe * mv;          // mv==0 for tail lanes -> no pollution
            p += __shfl_xor(p, 1, 32);
            p += __shfl_xor(p, 2, 32);
            p += __shfl_xor(p, 4, 32);
            p += __shfl_xor(p, 8, 32);       // sum over the 16 columns of this half
            if (c == 0) snorm[wave][nd] = p; // unique (wave, nd) writer
        }
        if (vok) {
            float* o = out + ((size_t)b * V + v) * ND + t * 16 + hi * 8;
            reinterpret_cast<float4*>(o)[0] = make_float4(vals[0], vals[1], vals[2], vals[3]);
            reinterpret_cast<float4*>(o)[1] = make_float4(vals[4], vals[5], vals[6], vals[7]);
        }
    }
    __syncthreads();
    if (tid < ND) {
        float s = 0.0f;
#pragma unroll
        for (int w = 0; w < 8; ++w) s += snorm[w][tid];
        atomicAdd(&normacc[b * ND + tid], s);
    }
}

// ---------------------------------------------------------------------------
// Kernel 3: in-place rescale of out (b,v,n) by rsqrt(feat_norm[b,n] + EPS)
// ---------------------------------------------------------------------------
__global__ void wks_norm(float* __restrict__ out,
                         const float* __restrict__ normacc,
                         int V, int total4) {
    const int i = blockIdx.x * blockDim.x + threadIdx.x;
    if (i >= total4) return;
    const int nq  = i & (ND / 4 - 1);       // quad within descriptor row (ND=128 -> 32)
    const int row = i >> 5;                 // b*V + v
    const int b   = row / V;
    const float* nr = normacc + b * ND + nq * 4;
    float4 q = reinterpret_cast<float4*>(out)[i];
    q.x *= rsqrtf(nr[0] + EPSF);
    q.y *= rsqrtf(nr[1] + EPSF);
    q.z *= rsqrtf(nr[2] + EPSF);
    q.w *= rsqrtf(nr[3] + EPSF);
    reinterpret_cast<float4*>(out)[i] = q;
}

// ---------------------------------------------------------------------------
extern "C" void kernel_launch(void* const* d_in, const int* in_sizes, int n_in,
                              void* d_out, int out_size, void* d_ws, size_t ws_size,
                              hipStream_t stream) {
    const float* evals = (const float*)d_in[0];
    const float* evecs = (const float*)d_in[1];
    const float* mass  = (const float*)d_in[2];
    const int K = in_sizes[0] / BATCH;      // 200
    const int V = in_sizes[2] / BATCH;      // 100000

    // workspace: bf16 coefs | sscale | normacc  (~234 KB total)
    unsigned short* coefsb = (unsigned short*)d_ws;
    float* sscale  = (float*)((char*)d_ws + (size_t)BATCH * ND * KPAD * sizeof(unsigned short));
    float* normacc = sscale + BATCH * ND;
    float* out = (float*)d_out;

    wks_prep<<<1, 256, 0, stream>>>(evals, coefsb, sscale, normacc, K);

    dim3 grid((V + 127) / 128, BATCH);
    wks_gemm<<<grid, 256, 0, stream>>>(evecs, mass, coefsb, sscale, normacc, out, V, K);

    const int total4 = BATCH * V * (ND / 4);
    wks_norm<<<(total4 + 255) / 256, 256, 0, stream>>>(out, normacc, V, total4);
}
